// RelMultiHeadAttn_19619410608620
// MI455X (gfx1250) — matrix operand order repaired and checked by reading, not compiled, not run on hardware
//
#include <hip/hip_runtime.h>
#include <hip/hip_bf16.h>

#define QLEN 1024
#define BSZ  4
#define DM   1024
#define NH   16
#define DH   64
#define RLEN 1025

typedef __attribute__((ext_vector_type(16))) __bf16 v16bf;
typedef __attribute__((ext_vector_type(8)))  __bf16 v8bf;
typedef __attribute__((ext_vector_type(8)))  float  v8f;
typedef __attribute__((ext_vector_type(4)))  float  v4f;

static __device__ __forceinline__ v8f wmma_bf16(v16bf a, v16bf b, v8f c) {
  // D = A(16x32 bf16) x B(32x16 bf16) + C(16x16 f32)
  return __builtin_amdgcn_wmma_f32_16x16x32_bf16(false, a, false, b, (short)0, c, false, false);
}
static __device__ __forceinline__ v8f v8f_zero() {
  v8f z = {0.f, 0.f, 0.f, 0.f, 0.f, 0.f, 0.f, 0.f};
  return z;
}
static __device__ __forceinline__ v16bf combine8(v8bf lo, v8bf hi) {
  return __builtin_shufflevector(lo, hi, 0, 1, 2, 3, 4, 5, 6, 7, 8, 9, 10, 11, 12, 13, 14, 15);
}
// A fragment (16x32 bf16, ISA 7.12.2): per lane two contiguous 8-elem runs at
// k = 8*half and k = 8*half + 16.  p points at k=0 of the 32-wide slice.
static __device__ __forceinline__ v16bf frag_a(const __bf16* p, int half) {
  const __bf16* q = p + 8 * half;
  return combine8(*(const v8bf*)q, *(const v8bf*)(q + 16));
}
// B fragment (32x16 bf16): per lane two contiguous 8-elem runs at
// k = 8*half and k = 8*half + 16 (same order as A).
static __device__ __forceinline__ v16bf frag_b(const __bf16* p, int half) {
  const __bf16* q = p + 8 * half;
  return combine8(*(const v8bf*)q, *(const v8bf*)(q + 16));
}
// CDNA5 async global->LDS copy (16B per lane), tracked with ASYNCcnt.
static __device__ __forceinline__ void async_load_b128(unsigned lds_off, const void* gaddr) {
  asm volatile("global_load_async_to_lds_b128 %0, %1, off"
               :: "v"(lds_off), "v"((unsigned long long)(size_t)gaddr)
               : "memory");
}
static __device__ __forceinline__ void wait_async0() {
  asm volatile("s_wait_asynccnt 0x0" ::: "memory");
}
static __device__ __forceinline__ void wait_ds0() {
  asm volatile("s_wait_dscnt 0x0" ::: "memory");
}

// ---------------------------------------------------------------------------
// Generic bf16 WMMA GEMM: C = A(MxK f32|bf16) x B(KxN f32), K = 1024.
// BM=64, BN=64, BK=64, 128 threads (4 waves in 2x2), each wave 32x32.
// B is transposed into LDS so both operands fragment-load as ds_load_b128.
// MODE 0: A=r, epilogue -> rq_bias (row RLEN-1, q part) + Rk bf16 (k part)
// MODE 1: A=w, epilogue -> scatter Qh(+bias)/Kh/Vh bf16 per (b,head)
// MODE 2: A=ctx (bf16), epilogue -> yout = C + wres (residual), f32
// ---------------------------------------------------------------------------
template <int MODE>
__global__ __launch_bounds__(128) void gemm_bf16_kernel(
    const void* __restrict__ Aptr, const float* __restrict__ Bptr,
    int M, int ldb,
    float* __restrict__ rq_bias, __bf16* __restrict__ Rk,
    __bf16* __restrict__ Qh, __bf16* __restrict__ Kh, __bf16* __restrict__ Vh,
    const float* __restrict__ wres, float* __restrict__ yout)
{
  const int K = 1024;
  __shared__ __align__(16) __bf16 At[64][72];   // 144B stride (16B multiple)
  __shared__ __align__(16) __bf16 BtT[64][72];  // B transposed: [n][k]

  const int tid  = threadIdx.x;
  const int lane = tid & 31;
  const int wid  = tid >> 5;
  const int r    = lane & 15;
  const int half = lane >> 4;
  const int bm0  = blockIdx.x * 64, bn0 = blockIdx.y * 64;
  const int wm   = (wid >> 1) * 32, wn = (wid & 1) * 32;

  v8f acc[2][2];
#pragma unroll
  for (int i = 0; i < 2; ++i)
#pragma unroll
    for (int j = 0; j < 2; ++j) acc[i][j] = v8f_zero();

  const int ar = tid >> 1, acol = (tid & 1) * 32;   // A tile: 64 rows x 64
  const int bkr = tid >> 1, bcb = (tid & 1) * 32;   // B tile: 64 k-rows x 64 n

  for (int k0 = 0; k0 < K; k0 += 64) {
    __syncthreads();
    {  // A tile -> LDS (bf16), vector 16B stores
      int gm = bm0 + ar;
      v8bf t[4];
      if (MODE == 2) {
        const __bf16* A = (const __bf16*)Aptr + (long)gm * K + k0 + acol;
#pragma unroll
        for (int c = 0; c < 4; ++c) t[c] = *(const v8bf*)(A + 8 * c);
      } else {
        const float* Afl = (const float*)Aptr;
        if (gm < M) {
          const float* ap = Afl + (long)gm * K + k0 + acol;
#pragma unroll
          for (int c = 0; c < 4; ++c)
#pragma unroll
            for (int i = 0; i < 8; ++i) t[c][i] = (__bf16)ap[8 * c + i];
        } else {
#pragma unroll
          for (int c = 0; c < 4; ++c)
#pragma unroll
            for (int i = 0; i < 8; ++i) t[c][i] = (__bf16)0.f;
        }
      }
#pragma unroll
      for (int c = 0; c < 4; ++c) *(v8bf*)&At[ar][acol + 8 * c] = t[c];
    }
    {  // B tile -> LDS transposed [n][k]
      const float* bp = Bptr + (long)(k0 + bkr) * ldb + bn0 + bcb;
#pragma unroll
      for (int i = 0; i < 32; ++i) BtT[bcb + i][bkr] = (__bf16)bp[i];
    }
    __syncthreads();

#pragma unroll
    for (int s = 0; s < 2; ++s) {
      v16bf a[2], b[2];
#pragma unroll
      for (int i = 0; i < 2; ++i) a[i] = frag_a(&At[wm + i * 16 + r][s * 32], half);
#pragma unroll
      for (int j = 0; j < 2; ++j) b[j] = frag_b(&BtT[wn + j * 16 + r][s * 32], half);
#pragma unroll
      for (int i = 0; i < 2; ++i)
#pragma unroll
        for (int j = 0; j < 2; ++j) acc[i][j] = wmma_bf16(a[i], b[j], acc[i][j]);
    }
  }

  // epilogue: C element (m = idx + 8*half, n = lane&15) per ISA C-layout
#pragma unroll
  for (int i = 0; i < 2; ++i) {
#pragma unroll
    for (int j = 0; j < 2; ++j) {
#pragma unroll
      for (int idx = 0; idx < 8; ++idx) {
        int gm = bm0 + wm + i * 16 + idx + 8 * half;
        int gn = bn0 + wn + j * 16 + r;
        float v = acc[i][j][idx];
        if (MODE == 0) {
          if (gn < 1024) {
            if (gm == RLEN - 1) rq_bias[gn] = v;  // r_head_q[-1]
          } else {
            int c = gn - 1024;                    // r_head_k
            if (gm < RLEN) Rk[((long)(c >> 6) * RLEN + gm) * DH + (c & 63)] = (__bf16)v;
          }
        } else if (MODE == 1) {
          int q = gm >> 2, bb2 = gm & 3;
          int part = gn >> 10, c = gn & 1023;
          int hh = c >> 6, dd = c & 63;
          long o = (((long)(bb2 * NH + hh) * QLEN) + q) * DH + dd;
          if (part == 0)      Qh[o] = (__bf16)(v + rq_bias[c]);  // rw_head_q
          else if (part == 1) Kh[o] = (__bf16)v;
          else                Vh[o] = (__bf16)v;
        } else {
          yout[(long)gm * DM + gn] = v + wres[(long)gm * DM + gn];
        }
      }
    }
  }
}

// ---------------------------------------------------------------------------
// Fused rel-shift flash attention. Block = (b, head, 64-query tile), 4 waves,
// each wave owns a 16-row query subtile with streaming softmax state.
// bd[i,j] = Q[i] . Rk[QLEN + j - i]  (elementwise rel_shift for j<=i) via an
// unshifted 16x80 band GEMM per subtile + transposed LDS spill + diagonal
// gather.  K/R tiles arrive via CDNA5 async global->LDS (ASYNCcnt); V is
// transposed into LDS at fill time so all WMMA operands load as ds_load_b128.
// ---------------------------------------------------------------------------
__global__ __launch_bounds__(128) void attn_kernel(
    const __bf16* __restrict__ Qh, const __bf16* __restrict__ Kh,
    const __bf16* __restrict__ Vh, const __bf16* __restrict__ Rk,
    __bf16* __restrict__ ctx)
{
  const int ib = blockIdx.x, h = blockIdx.y, b = blockIdx.z;
  const int i0 = ib * 64;
  const int tid = threadIdx.x, wid = tid >> 5, lane = tid & 31;
  const int r = lane & 15, half = lane >> 4;
  const int is = i0 + wid * 16;
  const float scale = 0.125f;  // 1/sqrt(64)

  // 62.5 KB static LDS; Pb aliases Kt (Kt dead after ac GEMMs, rewritten only
  // after the next loop-top barrier).
  __shared__ __align__(16) char smem[62464];
  __bf16 (*Kt)[72]   = (__bf16(*)[72])smem;            // 64  rows,  9216 B
  __bf16 (*VtT)[72]  = (__bf16(*)[72])(smem + 9216);   // 64  rows (V^T: [d][j])
  __bf16 (*Rw)[72]   = (__bf16(*)[72])(smem + 18432);  // 128 rows, 18432 B
  float  (*BDbT)[20] = (float(*)[20])(smem + 36864);   // 320 rows (bd^T), 25600 B
  __bf16 (*Pb)[72]   = (__bf16(*)[72])smem;            // alias of Kt

  const __bf16* Qp = Qh + (long)(b * NH + h) * QLEN * DH;
  const __bf16* Kp = Kh + (long)(b * NH + h) * QLEN * DH;
  const __bf16* Vp = Vh + (long)(b * NH + h) * QLEN * DH;
  const __bf16* Rp = Rk + (long)h * RLEN * DH;

  v16bf qa[2];
#pragma unroll
  for (int s = 0; s < 2; ++s)
    qa[s] = frag_a(Qp + (long)(is + r) * DH + s * 32, half);

  v8f O[4];
#pragma unroll
  for (int nt = 0; nt < 4; ++nt) O[nt] = v8f_zero();
  float mrun[8], lrun[8];
#pragma unroll
  for (int idx = 0; idx < 8; ++idx) { mrun[idx] = -3e38f; lrun[idx] = 0.f; }

  const int njt = ib + 1;  // causal: keys 0 .. i0+63
  for (int jt = 0; jt < njt; ++jt) {
    const int j0 = jt * 64;
    __syncthreads();
    // K tile: 64x64 bf16 = 512 x 16B async chunks
    for (int x = tid; x < 512; x += 128) {
      int rr = x >> 3, cc = (x & 7) * 8;
      async_load_b128((unsigned)(size_t)&Kt[rr][cc], Kp + (long)(j0 + rr) * DH + cc);
    }
    // Rk window: 128 rows (clamped), 1024 x 16B async chunks
    const int jrb = QLEN + j0 - i0 - 63;
    for (int x = tid; x < 1024; x += 128) {
      int rr = x >> 3, cc = (x & 7) * 8;
      int gr = jrb + rr;
      gr = gr < 0 ? 0 : (gr > RLEN - 1 ? RLEN - 1 : gr);
      async_load_b128((unsigned)(size_t)&Rw[rr][cc], Rp + (long)gr * DH + cc);
    }
    // V tile: plain vector load, transposed store -> VtT[d][j]
    for (int x = tid; x < 512; x += 128) {
      int rr = x >> 3, cc = (x & 7) * 8;
      v8bf vv = *(const v8bf*)(Vp + (long)(j0 + rr) * DH + cc);
#pragma unroll
      for (int i = 0; i < 8; ++i) VtT[cc + i][rr] = vv[i];
    }
    wait_async0();
    __syncthreads();

    // ac = Q . K^T   (B[k=d][n=key] = Kt[key][d], contiguous per lane)
    v8f sc[4];
#pragma unroll
    for (int nt = 0; nt < 4; ++nt) sc[nt] = v8f_zero();
#pragma unroll
    for (int s = 0; s < 2; ++s) {
      v16bf kb[4];
#pragma unroll
      for (int nt = 0; nt < 4; ++nt) kb[nt] = frag_b(&Kt[nt * 16 + r][s * 32], half);
#pragma unroll
      for (int nt = 0; nt < 4; ++nt) sc[nt] = wmma_bf16(qa[s], kb[nt], sc[nt]);
    }

    // bd band: 5 n-tiles over the wave's 80-wide jr window, spilled transposed
    const int off = 48 - 16 * wid;  // wave window start inside Rw
    {
      v8f bd[5];
#pragma unroll
      for (int n2 = 0; n2 < 5; ++n2) bd[n2] = v8f_zero();
#pragma unroll
      for (int s = 0; s < 2; ++s) {
        v16bf rb[5];
#pragma unroll
        for (int n2 = 0; n2 < 5; ++n2)
          rb[n2] = frag_b(&Rw[off + n2 * 16 + r][s * 32], half);
#pragma unroll
        for (int n2 = 0; n2 < 5; ++n2) bd[n2] = wmma_bf16(qa[s], rb[n2], bd[n2]);
      }
      // BDbT[col][m]: two ds_store_b128 per fragment (m = 8*half + 0..7)
#pragma unroll
      for (int n2 = 0; n2 < 5; ++n2) {
        v4f lo = __builtin_shufflevector(bd[n2], bd[n2], 0, 1, 2, 3);
        v4f hi = __builtin_shufflevector(bd[n2], bd[n2], 4, 5, 6, 7);
        float* dst = &BDbT[wid * 80 + n2 * 16 + r][8 * half];
        *(v4f*)dst = lo;
        *(v4f*)(dst + 4) = hi;
      }
    }
    wait_ds0();  // same-wave LDS RAW (DS in-order; belt & braces)

    // combine (diagonal gather), causal mask, streaming softmax
    float p[4][8], vmax[8];
#pragma unroll
    for (int idx = 0; idx < 8; ++idx) vmax[idx] = -3e38f;
#pragma unroll
    for (int nt = 0; nt < 4; ++nt) {
#pragma unroll
      for (int idx = 0; idx < 8; ++idx) {
        int m = idx + 8 * half;
        int jg = j0 + nt * 16 + r;
        float v = (sc[nt][idx] + BDbT[wid * 80 + 15 + nt * 16 + r - m][m]) * scale;
        if (jg > is + m) v = -3e38f;
        p[nt][idx] = v;
        vmax[idx] = fmaxf(vmax[idx], v);
      }
    }
#pragma unroll
    for (int msk = 1; msk <= 8; msk <<= 1)
#pragma unroll
      for (int idx = 0; idx < 8; ++idx)
        vmax[idx] = fmaxf(vmax[idx], __shfl_xor(vmax[idx], msk, 32));
    float sOld[8], rsum[8];
#pragma unroll
    for (int idx = 0; idx < 8; ++idx) {
      float mn = fmaxf(mrun[idx], vmax[idx]);
      sOld[idx] = __expf(mrun[idx] - mn);
      mrun[idx] = mn;
      rsum[idx] = 0.f;
    }
#pragma unroll
    for (int nt = 0; nt < 4; ++nt) {
#pragma unroll
      for (int idx = 0; idx < 8; ++idx) {
        int m = idx + 8 * half;
        float pe = __expf(p[nt][idx] - mrun[idx]);
        rsum[idx] += pe;
        Pb[wid * 16 + m][nt * 16 + r] = (__bf16)pe;  // aliases Kt (dead here)
      }
    }
#pragma unroll
    for (int msk = 1; msk <= 8; msk <<= 1)
#pragma unroll
      for (int idx = 0; idx < 8; ++idx)
        rsum[idx] += __shfl_xor(rsum[idx], msk, 32);
#pragma unroll
    for (int idx = 0; idx < 8; ++idx) lrun[idx] = lrun[idx] * sOld[idx] + rsum[idx];
#pragma unroll
    for (int nt = 0; nt < 4; ++nt)
#pragma unroll
      for (int idx = 0; idx < 8; ++idx) O[nt][idx] *= sOld[idx];
    wait_ds0();

    // O += P . V   (A[m][j] = Pb, B[j][d] = VtT[d][j], both contiguous)
#pragma unroll
    for (int s = 0; s < 2; ++s) {
      v16bf pa = frag_a(&Pb[wid * 16 + r][s * 32], half);
      v16bf vb[4];
#pragma unroll
      for (int nt = 0; nt < 4; ++nt) vb[nt] = frag_b(&VtT[nt * 16 + r][s * 32], half);
#pragma unroll
      for (int nt = 0; nt < 4; ++nt) O[nt] = wmma_bf16(pa, vb[nt], O[nt]);
    }
  }

  // normalize + write context in w-row layout (row = q*BSZ + b)
#pragma unroll
  for (int nt = 0; nt < 4; ++nt) {
#pragma unroll
    for (int idx = 0; idx < 8; ++idx) {
      int m = idx + 8 * half;
      float v = O[nt][idx] / lrun[idx];
      ctx[((long)(is + m) * BSZ + b) * DM + h * DH + nt * 16 + r] = (__bf16)v;
    }
  }
}

// ---------------------------------------------------------------------------
// In-place LayerNorm over rows of d_out ([4096 x 1024] f32), gamma/beta.
// ---------------------------------------------------------------------------
__global__ __launch_bounds__(256) void ln_kernel(
    float* __restrict__ y, const float* __restrict__ gamma,
    const float* __restrict__ beta)
{
  const int row = blockIdx.x, t = threadIdx.x;
  float* p = y + (long)row * DM;
  float v[4], s = 0.f, ss = 0.f;
#pragma unroll
  for (int i = 0; i < 4; ++i) {
    v[i] = p[t + 256 * i];
    s += v[i];
    ss += v[i] * v[i];
  }
#pragma unroll
  for (int m = 16; m >= 1; m >>= 1) {
    s += __shfl_xor(s, m, 32);
    ss += __shfl_xor(ss, m, 32);
  }
  __shared__ float sh_s[8], sh_q[8];
  if ((t & 31) == 0) { sh_s[t >> 5] = s; sh_q[t >> 5] = ss; }
  __syncthreads();
  float S = 0.f, Q = 0.f;
#pragma unroll
  for (int i = 0; i < 8; ++i) { S += sh_s[i]; Q += sh_q[i]; }
  float mu = S * (1.f / DM);
  float var = Q * (1.f / DM) - mu * mu;
  float inv = rsqrtf(var + 1e-5f);
#pragma unroll
  for (int i = 0; i < 4; ++i) {
    int c = t + 256 * i;
    p[c] = (v[i] - mu) * inv * gamma[c] + beta[c];
  }
}

extern "C" void kernel_launch(void* const* d_in, const int* in_sizes, int n_in,
                              void* d_out, int out_size, void* d_ws, size_t ws_size,
                              hipStream_t stream) {
  (void)in_sizes; (void)n_in; (void)out_size; (void)ws_size;
  const float* w     = (const float*)d_in[0];
  const float* rvec  = (const float*)d_in[1];
  // d_in[2] attn_mask unused: causal mask (j > i) is computed in-kernel
  const float* W_qkv = (const float*)d_in[3];
  const float* W_o   = (const float*)d_in[4];
  const float* gamma = (const float*)d_in[5];
  const float* beta  = (const float*)d_in[6];
  float* out = (float*)d_out;

  char* ws = (char*)d_ws;
  size_t off = 0;
  auto take = [&](size_t bytes) {
    void* p = ws + off;
    off = (off + bytes + 255) & ~(size_t)255;
    return p;
  };
  float*  rq_bias = (float*)take((size_t)NH * DH * sizeof(float));    //   4 KB
  __bf16* Rk      = (__bf16*)take((size_t)NH * RLEN * DH * 2);        //   2 MB
  __bf16* Qh      = (__bf16*)take((size_t)BSZ * NH * QLEN * DH * 2);  //   8 MB
  __bf16* Kh      = (__bf16*)take((size_t)BSZ * NH * QLEN * DH * 2);  //   8 MB
  __bf16* Vh      = (__bf16*)take((size_t)BSZ * NH * QLEN * DH * 2);  //   8 MB
  __bf16* ctx     = (__bf16*)take((size_t)QLEN * BSZ * DM * 2);       //   8 MB

  // 1) r @ W_qkv (q,k parts) -> rq_bias + Rk
  gemm_bf16_kernel<0><<<dim3(17, 32), 128, 0, stream>>>(
      rvec, W_qkv, RLEN, 3 * DM, rq_bias, Rk, nullptr, nullptr, nullptr, nullptr, nullptr);
  // 2) w @ W_qkv -> Qh(+bias)/Kh/Vh
  gemm_bf16_kernel<1><<<dim3(64, 48), 128, 0, stream>>>(
      w, W_qkv, QLEN * BSZ, 3 * DM, rq_bias, nullptr, Qh, Kh, Vh, nullptr, nullptr);
  // 3) fused rel-shift attention -> ctx
  attn_kernel<<<dim3(16, 16, 4), 128, 0, stream>>>(Qh, Kh, Vh, Rk, ctx);
  // 4) ctx @ W_o + residual -> d_out
  gemm_bf16_kernel<2><<<dim3(64, 16), 128, 0, stream>>>(
      ctx, W_o, QLEN * BSZ, DM, nullptr, nullptr, nullptr, nullptr, nullptr, w, out);
  // 5) LayerNorm in place
  ln_kernel<<<dim3(QLEN * BSZ), 256, 0, stream>>>(out, gamma, beta);
}
